// ASTGCNBlock_34033320854241
// MI455X (gfx1250) — matrix operand order repaired and checked by reading, not compiled, not run on hardware
//
#include <hip/hip_runtime.h>

// ASTGCN block forward for MI455X (gfx1250, wave32, WMMA + TDM).
// Dense contractions -> v_wmma_f32_16x16x32_f16 (f16 operands, fp32 accum).
// Heaviest GEMM (Vs @ P, K=512) stages operand slabs into LDS with the
// Tensor Data Mover (tensor_load_to_lds + s_wait_tensorcnt).
// Sparse Cheb propagation uses global float atomics.
// Workspace requirement: ~60 MB fp32 scratch in d_ws.

#define B_ 4
#define N_ 512
#define F_ 64
#define T_ 16
#define E_ 16384
#define FC_ 64
#define FT_ 64

typedef __attribute__((ext_vector_type(16))) _Float16 v16h;
typedef __attribute__((ext_vector_type(8)))  float    v8f;
typedef __attribute__((ext_vector_type(4)))  unsigned int v4u;
typedef __attribute__((ext_vector_type(4)))  int      v4i;
typedef __attribute__((ext_vector_type(8)))  int      v8i;

__device__ __forceinline__ void atomAddF(float* p, float v) {
  __hip_atomic_fetch_add(p, v, __ATOMIC_RELAXED, __HIP_MEMORY_SCOPE_AGENT);
}
__device__ __forceinline__ float sigm(float x) { return 1.0f / (1.0f + __expf(-x)); }

// ---------------- WMMA fragment helpers (wave32) ----------------
// A frag (16x32 f16), lane l: m=l&15, h=l>>4:
//   elems [0..7]  = A[m][8h + 0..7],  elems [8..15] = A[m][16+8h + 0..7]
// B frag (32x16 f16), lane l: n=l&15, h=l>>4:
//   elems [0..7]  = B[8h + 0..7][n],  elems [8..15] = B[16+8h + 0..7][n]
// C/D (16x16 f32): lane l holds col=l&15, rows r+8*(l>>4), r=0..7.
template <typename LA>
__device__ __forceinline__ v16h frag_a(LA loadA) {
  const int lane = threadIdx.x & 31;
  const int h = lane >> 4, m = lane & 15;
  v16h a;
#pragma unroll
  for (int i = 0; i < 8; ++i) {
    a[i]     = (_Float16)loadA(m, 8 * h + i);
    a[i + 8] = (_Float16)loadA(m, 16 + 8 * h + i);
  }
  return a;
}
template <typename LB>
__device__ __forceinline__ v16h frag_b(LB loadB) {
  const int lane = threadIdx.x & 31;
  const int h = lane >> 4, n = lane & 15;
  v16h b;
#pragma unroll
  for (int i = 0; i < 16; ++i) b[i] = (_Float16)loadB(i < 8 ? 8 * h + i : 16 + 8 * h + (i - 8), n);
  return b;
}

template <typename LA, typename LB>
__device__ __forceinline__ v8f wmma_tile(LA loadA, LB loadB, int K, v8f acc) {
  for (int k0 = 0; k0 < K; k0 += 32) {
    v16h a = frag_a([&](int m, int k) { return loadA(m, k0 + k); });
    v16h b = frag_b([&](int k, int n) { return loadB(k0 + k, n); });
    acc = __builtin_amdgcn_wmma_f32_16x16x32_f16(false, a, false, b,
                                                 (short)0, acc, false, false);
  }
  return acc;
}

template <typename ST>
__device__ __forceinline__ void wmma_store(v8f acc, ST st) {
  const int lane = threadIdx.x & 31;
  const int h = lane >> 4;
  const int col = lane & 15;
#pragma unroll
  for (int r = 0; r < 8; ++r) st(r + 8 * h, col, acc[r]);
}

__device__ __forceinline__ int waveId() {
  return (int)(blockIdx.x * (blockDim.x >> 5) + (threadIdx.x >> 5));
}

// ---------------- Tensor Data Mover: 2D fp32 tile -> LDS ----------------
// Loads a tileY x tileX fp32 tile (row stride `strideX` elements) from a
// 512x512 fp32 tensor at `gptr` into LDS at byte offset `ldsAddr`,
// packed row-major tileY x tileX. D# fields per CDNA5 ISA ch.8.
__device__ __forceinline__ void tdm_load_2d(unsigned ldsAddr, const float* gptr,
                                            unsigned tileX, unsigned tileY,
                                            unsigned strideX) {
  unsigned long long ga = (unsigned long long)(size_t)gptr;
  v4u g0;
  g0[0] = 1u;                                            // count=1 (valid D#)
  g0[1] = ldsAddr;                                       // lds_addr (bytes)
  g0[2] = (unsigned)(ga & 0xFFFFFFFFu);                  // global_addr[31:0]
  g0[3] = (unsigned)((ga >> 32) & 0x01FFFFFFu)           // global_addr[56:32]
          | 0x80000000u;                                 // type=2 ("image")
  v8i g1;
  g1[0] = (int)(2u << 16);                               // data_size=4B
  g1[1] = (int)((N_ & 0xFFFFu) << 16);                   // tensor_dim0 lo16
  g1[2] = (int)((N_ >> 16) | ((N_ & 0xFFFFu) << 16));    // dim0 hi | dim1 lo
  g1[3] = (int)((N_ >> 16) | (tileX << 16));             // dim1 hi | tile_dim0
  g1[4] = (int)(tileY & 0xFFFFu);                        // tile_dim1, tile_dim2=0
  g1[5] = (int)strideX;                                  // tensor_dim0_stride lo32
  g1[6] = 0;                                             // stride hi | dim1_stride
  g1[7] = 0;
  v4i gz = {0, 0, 0, 0};
#if defined(__clang_major__) && (__clang_major__ >= 23)
  v8i gz8 = {0, 0, 0, 0, 0, 0, 0, 0};
  __builtin_amdgcn_tensor_load_to_lds(g0, g1, gz, gz, gz8, 0);
#else
  __builtin_amdgcn_tensor_load_to_lds(g0, g1, gz, gz, 0);
#endif
}

// ---------------- small kernels ----------------
__global__ void k_zero(float* p, int n) {
  int i = blockIdx.x * 256 + threadIdx.x;
  if (i < n) p[i] = 0.0f;
}

// A1[b,t,f] = sum_n X[b,n,f,t] * U1[n]
__global__ void k_a1(const float* __restrict__ X, const float* __restrict__ U1,
                     float* __restrict__ A1) {
  int i = blockIdx.x * 256 + threadIdx.x;
  if (i >= B_ * T_ * F_) return;
  int b = i / (T_ * F_), t = (i / F_) % T_, f = i % F_;
  float s = 0.f;
  for (int n = 0; n < N_; ++n) s += X[((b * N_ + n) * F_ + f) * T_ + t] * U1[n];
  A1[i] = s;
}

// rhs_t[b,n,t] = sum_f U3[f] * X[b,n,f,t]
__global__ void k_rhs_t(const float* __restrict__ X, const float* __restrict__ U3,
                        float* __restrict__ rhs) {
  int i = blockIdx.x * 256 + threadIdx.x;
  if (i >= B_ * N_ * T_) return;
  int b = i / (N_ * T_), n = (i / T_) % N_, t = i % T_;
  float s = 0.f;
  for (int f = 0; f < F_; ++f) s += U3[f] * X[((b * N_ + n) * F_ + f) * T_ + t];
  rhs[i] = s;
}

// lhs_t[b,t,n] = sum_f A1[b,t,f] * U2[f,n]
__global__ void k_lhs_t(const float* __restrict__ A1, const float* __restrict__ U2,
                        float* __restrict__ lhs) {
  int i = blockIdx.x * 256 + threadIdx.x;
  if (i >= B_ * T_ * N_) return;
  int b = i / (T_ * N_), t = (i / N_) % T_, n = i % N_;
  float s = 0.f;
  for (int f = 0; f < F_; ++f) s += A1[(b * T_ + t) * F_ + f] * U2[f * N_ + n];
  lhs[i] = s;
}

// Et[b] = softmax_axis1( Ve @ sigmoid(lhs_t @ rhs_t + be) )   (one block per b)
__global__ void k_et(const float* __restrict__ lhs, const float* __restrict__ rhs,
                     const float* __restrict__ be, const float* __restrict__ Ve,
                     float* __restrict__ Et) {
  __shared__ float P[16][16];
  __shared__ float E0[16][16];
  int b = blockIdx.x;
  int t = threadIdx.x >> 4, u = threadIdx.x & 15;
  float s = 0.f;
  for (int n = 0; n < N_; ++n)
    s += lhs[(b * T_ + t) * N_ + n] * rhs[(b * N_ + n) * T_ + u];
  P[t][u] = sigm(s + be[t * T_ + u]);
  __syncthreads();
  float e = 0.f;
  for (int k = 0; k < T_; ++k) e += Ve[t * T_ + k] * P[k][u];
  E0[t][u] = e;
  __syncthreads();
  float m = -1e30f;
  for (int k = 0; k < T_; ++k) m = fmaxf(m, E0[k][u]);
  float z = 0.f;
  for (int k = 0; k < T_; ++k) z += __expf(E0[k][u] - m);
  Et[(b * T_ + t) * T_ + u] = __expf(E0[t][u] - m) / z;
}

// Xt = X.reshape(B, N*F, T) @ Et   (WMMA, K=16 padded to 32)
__global__ void k_xt(const float* __restrict__ X, const float* __restrict__ Et,
                     float* __restrict__ Xt) {
  int wid = waveId();
  if (wid >= B_ * (N_ * F_ / 16)) return;
  int b = wid / (N_ * F_ / 16);
  int m0 = (wid % (N_ * F_ / 16)) * 16;
  const float* Xb = X + (size_t)b * N_ * F_ * T_;
  const float* Eb = Et + b * T_ * T_;
  v8f acc = {};
  acc = wmma_tile(
      [&](int m, int k) { return k < T_ ? Xb[(size_t)(m0 + m) * T_ + k] : 0.f; },
      [&](int k, int n) { return k < T_ ? Eb[k * T_ + n] : 0.f; }, 32, acc);
  float* Ob = Xt + (size_t)b * N_ * F_ * T_;
  wmma_store(acc, [&](int r, int c, float v) { Ob[(size_t)(m0 + r) * T_ + c] = v; });
}

// A2[b,n,f] = sum_t Xt[b,n,f,t] * Ws1[t]
__global__ void k_a2(const float* __restrict__ Xt, const float* __restrict__ Ws1,
                     float* __restrict__ A2) {
  int i = blockIdx.x * 256 + threadIdx.x;
  if (i >= B_ * N_ * F_) return;
  const float* x = Xt + (size_t)i * T_;
  float s = 0.f;
  for (int t = 0; t < T_; ++t) s += x[t] * Ws1[t];
  A2[i] = s;
}

// lhs_s[b,n,t] = sum_f A2[b,n,f] * Ws2[f,t]
__global__ void k_lhs_s(const float* __restrict__ A2, const float* __restrict__ Ws2,
                        float* __restrict__ lhs) {
  int i = blockIdx.x * 256 + threadIdx.x;
  if (i >= B_ * N_ * T_) return;
  int bn = i / T_, t = i % T_;
  float s = 0.f;
  for (int f = 0; f < F_; ++f) s += A2[(size_t)bn * F_ + f] * Ws2[f * T_ + t];
  lhs[i] = s;
}

// rhs_s[b,t,n] = sum_f Ws3[f] * Xt[b,n,f,t]
__global__ void k_rhs_s(const float* __restrict__ Xt, const float* __restrict__ Ws3,
                        float* __restrict__ rhs) {
  int i = blockIdx.x * 256 + threadIdx.x;
  if (i >= B_ * T_ * N_) return;
  int b = i / (T_ * N_), t = (i / N_) % T_, n = i % N_;
  float s = 0.f;
  for (int f = 0; f < F_; ++f) s += Ws3[f] * Xt[((size_t)(b * N_ + n) * F_ + f) * T_ + t];
  rhs[i] = s;
}

// Ps[b,i,j] = sigmoid( sum_t lhs_s[b,i,t]*rhs_s[b,t,j] + bs[i,j] )   (WMMA K=16)
__global__ void k_ps(const float* __restrict__ lhs, const float* __restrict__ rhs,
                     const float* __restrict__ bs, float* __restrict__ Ps) {
  int wid = waveId();
  if (wid >= B_ * 32 * 32) return;
  int b = wid / 1024;
  int r0 = ((wid >> 5) & 31) * 16;
  int c0 = (wid & 31) * 16;
  v8f acc = {};
  acc = wmma_tile(
      [&](int m, int k) { return k < T_ ? lhs[((size_t)b * N_ + r0 + m) * T_ + k] : 0.f; },
      [&](int k, int n) { return k < T_ ? rhs[((size_t)b * T_ + k) * N_ + c0 + n] : 0.f; },
      32, acc);
  wmma_store(acc, [&](int r, int c, float v) {
    int i = r0 + r, j = c0 + c;
    Ps[((size_t)b * N_ + i) * N_ + j] = sigm(v + bs[(size_t)i * N_ + j]);
  });
}

// S[b] = Vs @ Ps[b]   (WMMA K=512; operand slabs staged into LDS by the TDM)
// Block = 128 threads (4 waves), computes a 64x64 output block.
// Per K-slab (32): TDM loads Vs[r0:r0+64, k0:k0+32] and Ps[b][k0:k0+32, c0:c0+64]
// into LDS; each wave owns a 16-row strip and sweeps 4 column tiles,
// reusing its A fragment across them.
__global__ void k_smm_tdm(const float* __restrict__ Vs, const float* __restrict__ Ps,
                          float* __restrict__ S) {
  __shared__ float lA[64 * 32];   // row-major 64 rows x 32 k
  __shared__ float lB[32 * 64];   // row-major 32 k  x 64 cols
  int bi = blockIdx.x;
  int b = bi >> 6;
  int r0 = ((bi >> 3) & 7) * 64;
  int c0 = (bi & 7) * 64;
  int w = threadIdx.x >> 5;
  const float* Psb = Ps + (size_t)b * N_ * N_;
  unsigned lAaddr = (unsigned)(size_t)lA;   // low 32 bits of flat LDS addr = LDS offset
  unsigned lBaddr = (unsigned)(size_t)lB;
  v8f acc[4] = {{}, {}, {}, {}};
  for (int s = 0; s < N_ / 32; ++s) {
    int k0 = s * 32;
    __syncthreads();                        // previous slab fully consumed
    if (w == 0) {
      tdm_load_2d(lAaddr, Vs + (size_t)r0 * N_ + k0, 32, 64, N_);
      tdm_load_2d(lBaddr, Psb + (size_t)k0 * N_ + c0, 64, 32, N_);
      __builtin_amdgcn_s_wait_tensorcnt(0);
    }
    __syncthreads();                        // slab visible to all waves
    v16h a = frag_a([&](int m, int k) { return lA[(w * 16 + m) * 32 + k]; });
#pragma unroll
    for (int cc = 0; cc < 4; ++cc) {
      v16h bf = frag_b([&](int k, int n) { return lB[k * 64 + cc * 16 + n]; });
      acc[cc] = __builtin_amdgcn_wmma_f32_16x16x32_f16(false, a, false, bf,
                                                       (short)0, acc[cc], false, false);
    }
  }
#pragma unroll
  for (int cc = 0; cc < 4; ++cc) {
    wmma_store(acc[cc], [&](int r, int c, float v) {
      S[((size_t)b * N_ + r0 + w * 16 + r) * N_ + c0 + cc * 16 + c] = v;
    });
  }
}

// fallback dense version (not launched; kept for quick revert)
__global__ void k_smm(const float* __restrict__ Vs, const float* __restrict__ Ps,
                      float* __restrict__ S) {
  int wid = waveId();
  if (wid >= B_ * 32 * 32) return;
  int b = wid / 1024;
  int r0 = ((wid >> 5) & 31) * 16;
  int c0 = (wid & 31) * 16;
  v8f acc = {};
  acc = wmma_tile(
      [&](int m, int k) { return Vs[(size_t)(r0 + m) * N_ + k]; },
      [&](int k, int n) { return Ps[((size_t)b * N_ + k) * N_ + c0 + n]; }, N_, acc);
  wmma_store(acc, [&](int r, int c, float v) {
    S[((size_t)b * N_ + r0 + r) * N_ + c0 + c] = v;
  });
}

// softmax over axis=1 (rows i) of S[b,i,j], in place; one thread per (b,j)
__global__ void k_softmax_s(float* __restrict__ S) {
  int i = blockIdx.x * 256 + threadIdx.x;
  if (i >= B_ * N_) return;
  int b = i / N_, j = i % N_;
  float* col = S + (size_t)b * N_ * N_ + j;
  float m = -1e30f;
  for (int r = 0; r < N_; ++r) m = fmaxf(m, col[(size_t)r * N_]);
  float z = 0.f;
  for (int r = 0; r < N_; ++r) z += __expf(col[(size_t)r * N_] - m);
  float inv = 1.0f / z;
  for (int r = 0; r < N_; ++r) col[(size_t)r * N_] = __expf(col[(size_t)r * N_] - m) * inv;
}

// degree count over edge sources
__global__ void k_deg(const int* __restrict__ row, float* __restrict__ deg) {
  int e = blockIdx.x * 256 + threadIdx.x;
  if (e >= E_) return;
  atomAddF(&deg[row[e]], 1.0f);
}

__global__ void k_dis(const float* __restrict__ deg, float* __restrict__ dis) {
  int n = blockIdx.x * 256 + threadIdx.x;
  if (n >= N_) return;
  float d = deg[n];
  dis[n] = (d > 0.f) ? rsqrtf(fmaxf(d, 1.0f)) : 0.0f;
}

__global__ void k_wn(const int* __restrict__ row, const int* __restrict__ col,
                     const float* __restrict__ dis, float* __restrict__ wn) {
  int e = blockIdx.x * 256 + threadIdx.x;
  if (e >= E_) return;
  wn[e] = -dis[row[e]] * dis[col[e]];
}

// Tx0[b,t,n,f] = S[b,n,n] * X[b,n,f,t]
__global__ void k_tx0(const float* __restrict__ X, const float* __restrict__ S,
                      float* __restrict__ Tx0) {
  int i = blockIdx.x * 256 + threadIdx.x;
  if (i >= B_ * T_ * N_ * F_) return;
  int b = i / (T_ * N_ * F_);
  int n = (i / F_) % N_;
  int f = i % F_;
  int t = (i / (N_ * F_)) % T_;
  Tx0[i] = S[((size_t)b * N_ + n) * N_ + n] * X[((size_t)(b * N_ + n) * F_ + f) * T_ + t];
}

__global__ void k_negcopy(const float* __restrict__ src, float* __restrict__ dst, int n) {
  int i = blockIdx.x * 256 + threadIdx.x;
  if (i < n) dst[i] = -src[i];
}

// dst[b,t,row,:] += w * src[b,t,col,:] over real edges
// (the +1/-1 self-loop pairs in the reference cancel exactly, both for
//  att_norm (S[b,i,i]*(+1-1)) and for plain norm, so they are skipped)
// w = wn[e] * (S ? S[b,row,col] : 1) * scale
__global__ void k_prop(const float* __restrict__ src, float* __restrict__ dst,
                       const float* __restrict__ wn, const int* __restrict__ row,
                       const int* __restrict__ col, const float* __restrict__ S,
                       float scale) {
  int i = blockIdx.x * 256 + threadIdx.x;
  if (i >= B_ * E_ * T_) return;
  int b = i / (E_ * T_);
  int e = (i / T_) % E_;
  int t = i % T_;
  int r = row[e], c = col[e];
  float w = wn[e] * scale;
  if (S) w *= S[((size_t)b * N_ + r) * N_ + c];
  const float* s = src + ((size_t)(b * T_ + t) * N_ + c) * F_;
  float* d = dst + ((size_t)(b * T_ + t) * N_ + r) * F_;
  __builtin_prefetch(s, 0, 1);       // global_prefetch_b8 on the gather row
#pragma unroll 4
  for (int f = 0; f < F_; ++f) atomAddF(&d[f], w * s[f]);
}

// Xhat = relu( Tx0@W0 + Tx1@W1 + Tx2@W2 + b_cheb )   (WMMA, 3 x K=64)
__global__ void k_cheb(const float* __restrict__ Tx0, const float* __restrict__ Tx1,
                       const float* __restrict__ Tx2, const float* __restrict__ Wc,
                       const float* __restrict__ bc, float* __restrict__ Xhat) {
  int wid = waveId();
  if (wid >= (B_ * T_ * N_ / 16) * (FC_ / 16)) return;
  int r0 = (wid >> 2) * 16;
  int c0 = (wid & 3) * 16;
  v8f acc = {};
  const float* Tx[3] = {Tx0, Tx1, Tx2};
  for (int p = 0; p < 3; ++p) {
    const float* A = Tx[p];
    const float* Bw = Wc + (size_t)p * F_ * FC_;
    acc = wmma_tile(
        [&](int m, int k) { return A[(size_t)(r0 + m) * F_ + k]; },
        [&](int k, int n) { return Bw[(size_t)k * FC_ + c0 + n]; }, F_, acc);
  }
  wmma_store(acc, [&](int r, int c, float v) {
    Xhat[(size_t)(r0 + r) * FC_ + c0 + c] = fmaxf(v + bc[c0 + c], 0.f);
  });
}

// Z[b,t,n,o] = relu( sum_dt sum_c Xhat[b,t+dt-1,n,c]*Wt[o,c,0,dt] + bt[o]
//                  + sum_c X[b,n,c,t]*Wr[o,c] + br[o] )        (WMMA)
__global__ void k_tconv(const float* __restrict__ Xhat, const float* __restrict__ X,
                        const float* __restrict__ Wt, const float* __restrict__ bt,
                        const float* __restrict__ Wr, const float* __restrict__ br,
                        float* __restrict__ Z) {
  int wid = waveId();
  if (wid >= (B_ * T_ * N_ / 16) * (FT_ / 16)) return;
  int rt = wid >> 2;
  int c0 = (wid & 3) * 16;
  int b = rt / (T_ * (N_ / 16));
  int t = (rt / (N_ / 16)) % T_;
  int n0 = (rt % (N_ / 16)) * 16;
  v8f acc = {};
  for (int dt = 0; dt < 3; ++dt) {
    int ts = t + dt - 1;                 // zero padding on T
    if (ts < 0 || ts >= T_) continue;    // uniform per wave -> EXEC stays full
    const float* A = Xhat + ((size_t)(b * T_ + ts) * N_ + n0) * FC_;
    acc = wmma_tile(
        [&](int m, int k) { return A[(size_t)m * FC_ + k]; },
        [&](int k, int n) { return Wt[(size_t)(c0 + n) * (FC_ * 3) + k * 3 + dt]; },
        FC_, acc);
  }
  acc = wmma_tile(
      [&](int m, int k) { return X[((size_t)(b * N_ + n0 + m) * F_ + k) * T_ + t]; },
      [&](int k, int n) { return Wr[(size_t)(c0 + n) * F_ + k]; }, F_, acc);
  wmma_store(acc, [&](int r, int c, float v) {
    int o = c0 + c;
    Z[((size_t)(b * T_ + t) * N_ + n0 + r) * FT_ + o] = fmaxf(v + bt[o] + br[o], 0.f);
  });
}

// LayerNorm over FT + final transpose (B,T,N,Ft) -> (B,N,Ft,T)
__global__ void k_ln(const float* __restrict__ Z, const float* __restrict__ gamma,
                     const float* __restrict__ beta, float* __restrict__ out) {
  int i = blockIdx.x * 256 + threadIdx.x;
  if (i >= B_ * T_ * N_) return;
  int b = i / (T_ * N_);
  int t = (i / N_) % T_;
  int n = i % N_;
  const float* z = Z + (size_t)i * FT_;
  float mu = 0.f;
  for (int o = 0; o < FT_; ++o) mu += z[o];
  mu *= (1.0f / FT_);
  float var = 0.f;
  for (int o = 0; o < FT_; ++o) { float d = z[o] - mu; var += d * d; }
  var *= (1.0f / FT_);
  float rs = rsqrtf(var + 1e-5f);
  float* ob = out + (size_t)(b * N_ + n) * FT_ * T_ + t;
  for (int o = 0; o < FT_; ++o)
    ob[(size_t)o * T_] = (z[o] - mu) * rs * gamma[o] + beta[o];
}

// ---------------- workspace layout (fp32 elements) ----------------
constexpr size_t oA1  = 0;
constexpr size_t oLT  = oA1 + (size_t)B_ * T_ * F_;
constexpr size_t oRT  = oLT + (size_t)B_ * T_ * N_;
constexpr size_t oEt  = oRT + (size_t)B_ * N_ * T_;
constexpr size_t oXt  = oEt + (size_t)B_ * T_ * T_;
constexpr size_t oA2  = oXt + (size_t)B_ * N_ * F_ * T_;
constexpr size_t oLS  = oA2 + (size_t)B_ * N_ * F_;
constexpr size_t oRS  = oLS + (size_t)B_ * N_ * T_;
constexpr size_t oPs  = oRS + (size_t)B_ * T_ * N_;
constexpr size_t oS   = oPs + (size_t)B_ * N_ * N_;
constexpr size_t oDeg = oS + (size_t)B_ * N_ * N_;
constexpr size_t oDis = oDeg + (size_t)N_;
constexpr size_t oWn  = oDis + (size_t)N_;
constexpr size_t oT0  = oWn + (size_t)E_;
constexpr size_t oT1  = oT0 + (size_t)B_ * T_ * N_ * F_;
constexpr size_t oT2  = oT1 + (size_t)B_ * T_ * N_ * F_;
constexpr size_t oXh  = oT2 + (size_t)B_ * T_ * N_ * F_;
constexpr size_t oZ   = oXh + (size_t)B_ * T_ * N_ * FC_;

extern "C" void kernel_launch(void* const* d_in, const int* in_sizes, int n_in,
                              void* d_out, int out_size, void* d_ws, size_t ws_size,
                              hipStream_t stream) {
  (void)in_sizes; (void)n_in; (void)out_size; (void)ws_size;
  const float* X   = (const float*)d_in[0];
  const int*   ei  = (const int*)d_in[1];
  const int*   row = ei;
  const int*   col = ei + E_;
  const float* U1  = (const float*)d_in[2];
  const float* U2  = (const float*)d_in[3];
  const float* U3  = (const float*)d_in[4];
  const float* be  = (const float*)d_in[5];
  const float* Ve  = (const float*)d_in[6];
  const float* Ws1 = (const float*)d_in[7];
  const float* Ws2 = (const float*)d_in[8];
  const float* Ws3 = (const float*)d_in[9];
  const float* bs  = (const float*)d_in[10];
  const float* Vs  = (const float*)d_in[11];
  const float* Wc  = (const float*)d_in[12];
  const float* bc  = (const float*)d_in[13];
  const float* Wt  = (const float*)d_in[14];
  const float* bt  = (const float*)d_in[15];
  const float* Wr  = (const float*)d_in[16];
  const float* br  = (const float*)d_in[17];
  const float* gm  = (const float*)d_in[18];
  const float* bb  = (const float*)d_in[19];
  float* out = (float*)d_out;
  float* ws  = (float*)d_ws;

  const int nBTN  = B_ * T_ * N_;                         // 32768
  const int nBTNF = B_ * T_ * N_ * F_;                    // 2097152
  const int wTiles = (nBTN / 16) * (FC_ / 16);            // 8192 waves

  // ---- temporal attention ----
  k_a1   <<<(B_ * T_ * F_ + 255) / 256, 256, 0, stream>>>(X, U1, ws + oA1);
  k_rhs_t<<<(B_ * N_ * T_ + 255) / 256, 256, 0, stream>>>(X, U3, ws + oRT);
  k_lhs_t<<<(B_ * T_ * N_ + 255) / 256, 256, 0, stream>>>(ws + oA1, U2, ws + oLT);
  k_et   <<<B_, 256, 0, stream>>>(ws + oLT, ws + oRT, be, Ve, ws + oEt);
  k_xt   <<<(B_ * (N_ * F_ / 16) + 3) / 4, 128, 0, stream>>>(X, ws + oEt, ws + oXt);

  // ---- spatial attention ----
  k_a2   <<<(B_ * N_ * F_ + 255) / 256, 256, 0, stream>>>(ws + oXt, Ws1, ws + oA2);
  k_lhs_s<<<(B_ * N_ * T_ + 255) / 256, 256, 0, stream>>>(ws + oA2, Ws2, ws + oLS);
  k_rhs_s<<<(B_ * T_ * N_ + 255) / 256, 256, 0, stream>>>(ws + oXt, Ws3, ws + oRS);
  k_ps   <<<(B_ * 32 * 32 + 3) / 4, 128, 0, stream>>>(ws + oLS, ws + oRS, bs, ws + oPs);
  k_smm_tdm<<<B_ * 8 * 8, 128, 0, stream>>>(Vs, ws + oPs, ws + oS);
  k_softmax_s<<<(B_ * N_ + 255) / 256, 256, 0, stream>>>(ws + oS);

  // ---- graph normalization ----
  k_zero<<<(N_ + 255) / 256, 256, 0, stream>>>(ws + oDeg, N_);
  k_deg <<<(E_ + 255) / 256, 256, 0, stream>>>(row, ws + oDeg);
  k_dis <<<(N_ + 255) / 256, 256, 0, stream>>>(ws + oDeg, ws + oDis);
  k_wn  <<<(E_ + 255) / 256, 256, 0, stream>>>(row, col, ws + oDis, ws + oWn);

  // ---- Chebyshev recursion ----
  k_tx0<<<(nBTNF + 255) / 256, 256, 0, stream>>>(X, ws + oS, ws + oT0);
  k_zero<<<(nBTNF + 255) / 256, 256, 0, stream>>>(ws + oT1, nBTNF);
  k_prop<<<(B_ * E_ * T_ + 255) / 256, 256, 0, stream>>>(
      ws + oT0, ws + oT1, ws + oWn, row, col, ws + oS, 1.0f);
  k_negcopy<<<(nBTNF + 255) / 256, 256, 0, stream>>>(ws + oT0, ws + oT2, nBTNF);
  k_prop<<<(B_ * E_ * T_ + 255) / 256, 256, 0, stream>>>(
      ws + oT1, ws + oT2, ws + oWn, row, col, nullptr, 2.0f);
  k_cheb<<<(wTiles + 3) / 4, 128, 0, stream>>>(
      ws + oT0, ws + oT1, ws + oT2, Wc, bc, ws + oXh);

  // ---- time conv + residual + relu ----
  k_tconv<<<(wTiles + 3) / 4, 128, 0, stream>>>(ws + oXh, X, Wt, bt, Wr, br, ws + oZ);

  // ---- layernorm + output transpose ----
  k_ln<<<(nBTN + 255) / 256, 256, 0, stream>>>(ws + oZ, gm, bb, out);
}
